// RepFlowLayer_1898375545119
// MI455X (gfx1250) — matrix operand order repaired and checked by reading, not compiled, not run on hardware
//
#include <hip/hip_runtime.h>
#include <hip/hip_bf16.h>

typedef __attribute__((ext_vector_type(16))) _Float16 v16h;
typedef __attribute__((ext_vector_type(8)))  float    v8f;

// fast silu: single native v_rcp_f32 instead of IEEE divide sequence
__device__ __forceinline__ float silu_f(float x) {
    return x * __builtin_amdgcn_rcpf(1.0f + __expf(-x));
}

__device__ __forceinline__ v8f wmma_f32_f16(v16h a, v16h b, v8f c) {
    return __builtin_amdgcn_wmma_f32_16x16x32_f16(false, a, false, b, (short)0, c, false, false);
}

// ---- fragment loaders -------------------------------------------------------
// A (16x32, f16) from row-major f32: lane row = m0+(lane&15), K-halves by lane>>4.
// Two contiguous 8-float runs -> 4x b128 loads, then packed converts.
__device__ __forceinline__ v16h frag_a_f32(const float* base, int ld, int m0, int k0, int lane) {
    int m  = m0 + (lane & 15);
    int kh = (lane >> 4) << 3;           // 0 or 8
    const float* p = base + (long)m * ld + k0 + kh;
    float4 x0 = *(const float4*)(p);
    float4 x1 = *(const float4*)(p + 4);
    float4 x2 = *(const float4*)(p + 16);
    float4 x3 = *(const float4*)(p + 20);
    v16h a;
    a[0]  = (_Float16)x0.x; a[1]  = (_Float16)x0.y; a[2]  = (_Float16)x0.z; a[3]  = (_Float16)x0.w;
    a[4]  = (_Float16)x1.x; a[5]  = (_Float16)x1.y; a[6]  = (_Float16)x1.z; a[7]  = (_Float16)x1.w;
    a[8]  = (_Float16)x2.x; a[9]  = (_Float16)x2.y; a[10] = (_Float16)x2.z; a[11] = (_Float16)x2.w;
    a[12] = (_Float16)x3.x; a[13] = (_Float16)x3.y; a[14] = (_Float16)x3.z; a[15] = (_Float16)x3.w;
    return a;
}

// B (32x16, f16) from pre-swizzled fragment-ready weights:
// layout [kt][nt][lane][16 halfs] -> one aligned 32-byte load per lane.
__device__ __forceinline__ v16h frag_b_pk(const _Float16* Wp, int NT, int kt, int nt, int lane) {
    return ((const v16h*)Wp)[(kt * NT + nt) * 32 + lane];
}

// A fragment over concatenated [nei(128) | edge(64)] rows of one l (K=192).
__device__ __forceinline__ v16h frag_a_cat(const float* nei, const float* edge,
                                           int l, int m0, int k0, int lane) {
    int j  = m0 + (lane & 15);
    int kh = (lane >> 4) << 3;
    const float* p;
    if (k0 < 128) p = nei  + ((long)l * 64 + j) * 128 + k0 + kh;
    else          p = edge + ((long)l * 64 + j) * 64  + (k0 - 128) + kh;
    float4 x0 = *(const float4*)(p);
    float4 x1 = *(const float4*)(p + 4);
    float4 x2 = *(const float4*)(p + 16);
    float4 x3 = *(const float4*)(p + 20);
    v16h a;
    a[0]  = (_Float16)x0.x; a[1]  = (_Float16)x0.y; a[2]  = (_Float16)x0.z; a[3]  = (_Float16)x0.w;
    a[4]  = (_Float16)x1.x; a[5]  = (_Float16)x1.y; a[6]  = (_Float16)x1.z; a[7]  = (_Float16)x1.w;
    a[8]  = (_Float16)x2.x; a[9]  = (_Float16)x2.y; a[10] = (_Float16)x2.z; a[11] = (_Float16)x2.w;
    a[12] = (_Float16)x3.x; a[13] = (_Float16)x3.y; a[14] = (_Float16)x3.z; a[15] = (_Float16)x3.w;
    return a;
}

// ---- K0: pack weights f32 -> f16 in WMMA-B fragment layout ------------------
// out element g: tile f = g>>9, r = g&511, lane = r>>4, e = r&15
//   kt = f / (N/16), nt = f % (N/16)
//   logical (k, n) = (kt*32 + (lane>>4)*8 + (e&7) + (e>>3)*16, nt*16 + (lane&15))
// SRC 0: S0[(KOFF+k)*N + n]               (ld == N)
// SRC 2: col-split at 64, both ld 64:     (n<64) ? S0[(KOFF+k)*64+n] : S1[(KOFF+k)*64+n-64]
// SRC 3: quad split (A2|A3|B2|B3):        sec=n>>6, row off 192/256, S0/S1
template <int K, int N, int SRC, int KOFF>
__global__ void __launch_bounds__(256) pack_weight(
        const float* __restrict__ S0, const float* __restrict__ S1,
        _Float16* __restrict__ out) {
    constexpr int NT  = N / 16;
    constexpr int TOT = (K / 32) * NT * 512;
    int g = blockIdx.x * 256 + threadIdx.x;
    if (g >= TOT) return;
    int f = g >> 9, r = g & 511;
    int lane = r >> 4, e = r & 15;
    int kt = f / NT, nt = f - kt * NT;
    int kk = kt * 32 + ((lane >> 4) << 3) + (e & 7) + ((e >> 3) << 4);
    int n  = nt * 16 + (lane & 15);
    float v;
    if constexpr (SRC == 0) {
        v = S0[(KOFF + kk) * N + n];
    } else if constexpr (SRC == 2) {
        v = (n < 64) ? S0[(KOFF + kk) * 64 + n] : S1[(KOFF + kk) * 64 + (n - 64)];
    } else { // SRC == 3
        int sec = n >> 6, nn = n & 63;
        const float* P = (sec < 2) ? S0 : S1;
        int ro = 192 + (sec & 1) * 64;
        v = P[(ro + kk) * 64 + nn];
    }
    out[g] = (_Float16)v;
}

// ---- K1: gather nei_node ----------------------------------------------------
__global__ void __launch_bounds__(256) gather_nei(
        const float* __restrict__ node_ext, const int* __restrict__ nlist,
        const unsigned char* __restrict__ msk, float* __restrict__ nei) {
    long g  = (long)blockIdx.x * 256 + threadIdx.x;   // 1024*64*128
    int  d  = (int)(g & 127);
    long lj = g >> 7;
    int idx = msk[lj] ? nlist[lj] : 0;
    nei[g] = node_ext[(long)idx * 128 + d];
}

// ---- K3: grrg (symmetrization features) -> concat768 ------------------------
__global__ void __launch_bounds__(256) grrg_kernel(
        const float* __restrict__ h2, const float* __restrict__ edge,
        const float* __restrict__ nei, const float* __restrict__ sw,
        const unsigned char* __restrict__ msk, float* __restrict__ concat) {
    int l = blockIdx.x, tid = threadIdx.x;
    __shared__ float hg_e[3][64];
    __shared__ float hg_n[3][128];
    __shared__ float swm[64];
    __shared__ float h2s[64][3];
    if (tid < 64)  swm[tid] = msk[l * 64 + tid] ? sw[l * 64 + tid] : 0.0f;
    if (tid < 192) { int n = tid / 3, t = tid % 3; h2s[n][t] = h2[((long)l * 64 + n) * 3 + t]; }
    __syncthreads();
    for (int idx = tid; idx < 192; idx += 256) {
        int t = idx / 64, d = idx & 63; float s = 0.0f;
        for (int n = 0; n < 64; ++n) s += h2s[n][t] * edge[((long)l * 64 + n) * 64 + d] * swm[n];
        hg_e[t][d] = s * (1.0f / 64.0f);
    }
    for (int idx = tid; idx < 384; idx += 256) {
        int t = idx / 128, d = idx & 127; float s = 0.0f;
        for (int n = 0; n < 64; ++n) s += h2s[n][t] * nei[((long)l * 64 + n) * 128 + d] * swm[n];
        hg_n[t][d] = s * (1.0f / 64.0f);
    }
    __syncthreads();
    float* crow = concat + (long)l * 768;
    for (int idx = tid; idx < 256; idx += 256) {
        int a = idx >> 6, d = idx & 63;
        crow[idx] = (hg_e[0][a] * hg_e[0][d] + hg_e[1][a] * hg_e[1][d] + hg_e[2][a] * hg_e[2][d]) * (1.0f / 3.0f);
    }
    for (int idx = tid; idx < 512; idx += 256) {
        int a = idx >> 7, d = idx & 127;
        crow[256 + idx] = (hg_n[0][a] * hg_n[0][d] + hg_n[1][a] * hg_n[1][d] + hg_n[2][a] * hg_n[2][d]) * (1.0f / 3.0f);
    }
}

// ---- K2: node-level GEMMs, templated so K/N/activation are compile-time -----
template <int JOB>
__global__ void __launch_bounds__(256) node_gemm_t(
        const float* __restrict__ A, const _Float16* __restrict__ Wp,
        const float* __restrict__ bias, float* __restrict__ out) {
    constexpr int  K   = (JOB == 1) ? 768 : 128;
    constexpr int  N   = (JOB == 4) ? 64 : 128;
    constexpr int  NT  = N / 16;
    constexpr bool ACT = (JOB == 0 || JOB == 1);
    int m0 = blockIdx.x * 16;
    int lane = threadIdx.x & 31, wv = threadIdx.x >> 5;
    int n0 = wv * 16;
    if (n0 >= N) return;                       // whole-wave uniform exit
    int nt = wv;
    v8f c = {};
#pragma unroll
    for (int kt = 0; kt < K / 32; ++kt) {
        v16h a = frag_a_f32(A, K, m0, kt * 32, lane);
        v16h b = frag_b_pk(Wp, NT, kt, nt, lane);
        c = wmma_f32_f16(a, b, c);
    }
    int col = n0 + (lane & 15);
    int rbase = m0 + ((lane >> 4) << 3);
    float bv = ACT ? bias[col] : 0.0f;
#pragma unroll
    for (int r = 0; r < 8; ++r) {
        float v = c[r];
        if (ACT) v = silu_f(v + bv);
        out[(long)(rbase + r) * N + col] = v;
    }
}

// ---- K2b: e_ang @ [A2|A3|B2|B3] (20480 x 64 x 256) --------------------------
__global__ void __launch_bounds__(256) eab_gemm(
        const float* __restrict__ edge, const unsigned char* __restrict__ amask,
        const _Float16* __restrict__ A23B23, float* __restrict__ eAB) {
    int m0 = blockIdx.x * 16;
    int lane = threadIdx.x & 31, wv = threadIdx.x >> 5;
    int nt = blockIdx.y * 8 + wv;
    int row = m0 + (lane & 15);
    int l = row / 20, j = row - l * 20;
    float mv = amask[row] ? 1.0f : 0.0f;
    const float* prow = edge + ((long)l * 64 + j) * 64 + ((lane >> 4) << 3);
    v8f c = {};
#pragma unroll
    for (int kt = 0; kt < 2; ++kt) {
        const float* p = prow + kt * 32;
        float4 x0 = *(const float4*)(p);
        float4 x1 = *(const float4*)(p + 4);
        float4 x2 = *(const float4*)(p + 16);
        float4 x3 = *(const float4*)(p + 20);
        v16h a;
        a[0]  = (_Float16)(x0.x * mv); a[1]  = (_Float16)(x0.y * mv);
        a[2]  = (_Float16)(x0.z * mv); a[3]  = (_Float16)(x0.w * mv);
        a[4]  = (_Float16)(x1.x * mv); a[5]  = (_Float16)(x1.y * mv);
        a[6]  = (_Float16)(x1.z * mv); a[7]  = (_Float16)(x1.w * mv);
        a[8]  = (_Float16)(x2.x * mv); a[9]  = (_Float16)(x2.y * mv);
        a[10] = (_Float16)(x2.z * mv); a[11] = (_Float16)(x2.w * mv);
        a[12] = (_Float16)(x3.x * mv); a[13] = (_Float16)(x3.y * mv);
        a[14] = (_Float16)(x3.z * mv); a[15] = (_Float16)(x3.w * mv);
        v16h b = frag_b_pk(A23B23, 16, kt, nt, lane);
        c = wmma_f32_f16(a, b, c);
    }
    int col = nt * 16 + (lane & 15);
    int rbase = m0 + ((lane >> 4) << 3);
#pragma unroll
    for (int r = 0; r < 8; ++r) eAB[(long)(rbase + r) * 256 + col] = c[r];
}

// ---- K5: fused angle kernel: angle @ [A0|B0], both epilogues ----------------
__global__ void __launch_bounds__(256) angle_main(
        const float* __restrict__ angle, const float* __restrict__ a_sw,
        const unsigned char* __restrict__ amask,
        const float* __restrict__ nAB1, const float* __restrict__ eAB,
        const float* __restrict__ b_ea1, const float* __restrict__ b_as,
        const float* __restrict__ res_a, const _Float16* __restrict__ A0B0,
        float* __restrict__ padded, float* __restrict__ out_angle) {
    int l = blockIdx.y;
    int m0 = blockIdx.x * 16;                       // 400 rows = (i,j)
    int lane = threadIdx.x & 31, wv = threadIdx.x >> 5;
    int nt = wv;                                    // 8 tiles: [pre_ang | angle_self]
    const float* Al = angle + (long)l * 400 * 64;
    v8f c = {};
#pragma unroll
    for (int kt = 0; kt < 2; ++kt) {
        v16h a = frag_a_f32(Al, 64, m0, kt * 32, lane);
        v16h b = frag_b_pk(A0B0, 8, kt, nt, lane);
        c = wmma_f32_f16(a, b, c);
    }
    int col = nt * 16 + (lane & 15);
    int rbase = m0 + ((lane >> 4) << 3);
    const float* eABl = eAB + (long)l * 20 * 256;
    if (col < 64) {                                 // pre_ang -> silu -> a_sw weighted j-sum
        int d = col;
        float nadd = nAB1[l * 128 + d] + b_ea1[d];
        const float inv = 0.22360679774997896f;     // 1/sqrt(20)
        float acc = 0.0f, aswi = 0.0f; int cur_i = -1;
#pragma unroll
        for (int r = 0; r < 8; ++r) {
            int row = rbase + r;
            int i = row / 20, jj = row - i * 20;
            float val = c[r] + nadd + eABl[jj * 256 + d] + eABl[i * 256 + 64 + d];
            float s = silu_f(val);
            float aswj = amask[l * 20 + jj] ? a_sw[l * 20 + jj] : 0.0f;
            if (i != cur_i) {
                if (cur_i >= 0) atomicAdd(&padded[((long)l * 64 + cur_i) * 64 + d], acc);
                cur_i = i; acc = 0.0f;
                aswi = amask[l * 20 + i] ? a_sw[l * 20 + i] : 0.0f;
            }
            acc += aswi * aswj * inv * s;
        }
        if (cur_i >= 0) atomicAdd(&padded[((long)l * 64 + cur_i) * 64 + d], acc);
    } else {                                        // angle_self -> angle_new
        int d = col - 64;
        float nadd = nAB1[l * 128 + 64 + d] + b_as[d];
        float ra = res_a[d];
#pragma unroll
        for (int r = 0; r < 8; ++r) {
            int row = rbase + r;
            int i = row / 20, jj = row - i * 20;
            float val = c[r] + nadd + eABl[jj * 256 + 128 + d] + eABl[i * 256 + 192 + d];
            long off = ((long)l * 400 + row) * 64 + d;
            out_angle[off] = angle[off] + ra * silu_f(val);
        }
    }
}

// ---- K4: fused [nei|edge] @ [W1;W2] and @ [V1;V2]; node_new + edge base -----
__global__ void __launch_bounds__(256) ne_es_kernel(
        const float* __restrict__ node, const float* __restrict__ edge,
        const float* __restrict__ nei, const float* __restrict__ sw,
        const unsigned char* __restrict__ msk,
        const float* __restrict__ nW0, const float* __restrict__ nV0,
        const float* __restrict__ n_self, const float* __restrict__ node_sym,
        const float* __restrict__ b_ne, const float* __restrict__ b_es,
        const float* __restrict__ res_n, const float* __restrict__ res_e,
        const _Float16* __restrict__ WneCat, const _Float16* __restrict__ WesCat,
        float* __restrict__ out_node, float* __restrict__ out_edge) {
    int l = blockIdx.x;
    int tid = threadIdx.x, lane = tid & 31, wv = tid >> 5;
    __shared__ float sh_nedge[128];
    __shared__ float sh_sw[64];
    if (tid < 128) sh_nedge[tid] = 0.0f;
    if (tid < 64)  sh_sw[tid] = msk[l * 64 + tid] ? sw[l * 64 + tid] : 0.0f;
    __syncthreads();
    for (int t = wv; t < 48; t += 8) {
        bool is_ne = t < 32;
        int mt, nt, NT; const _Float16* W;
        if (is_ne) { mt = t >> 3; nt = t & 7; NT = 8; W = WneCat; }
        else       { int t2 = t - 32; mt = t2 >> 2; nt = t2 & 3; NT = 4; W = WesCat; }
        int m0 = mt * 16, n0 = nt * 16;
        v8f c = {};
#pragma unroll
        for (int kt = 0; kt < 6; ++kt) {
            v16h a = frag_a_cat(nei, edge, l, m0, kt * 32, lane);
            v16h b = frag_b_pk(W, NT, kt, nt, lane);
            c = wmma_f32_f16(a, b, c);
        }
        int col = n0 + (lane & 15);
        int rbase = m0 + ((lane >> 4) << 3);
        if (is_ne) {
            float add = nW0[l * 128 + col] + b_ne[col];
            float part = 0.0f;
#pragma unroll
            for (int r = 0; r < 8; ++r) {
                int j = rbase + r;
                part += silu_f(c[r] + add) * sh_sw[j];
            }
            atomicAdd(&sh_nedge[col], part);
        } else {
            float add = nV0[l * 64 + col] + b_es[col];
            float r0 = res_e[col];
#pragma unroll
            for (int r = 0; r < 8; ++r) {
                int j = rbase + r;
                long off = ((long)l * 64 + j) * 64 + col;
                out_edge[off] = edge[off] + r0 * silu_f(c[r] + add);
            }
        }
    }
    __syncthreads();
    if (tid < 128) {
        int d = tid;
        float ne = sh_nedge[d] * (1.0f / 64.0f);
        out_node[l * 128 + d] = node[l * 128 + d] + res_n[d] * n_self[l * 128 + d]
                              + res_n[128 + d] * node_sym[l * 128 + d] + res_n[256 + d] * ne;
    }
}

// ---- K6: padded @ W_ea2 -> silu -> accumulate into edge_new -----------------
__global__ void __launch_bounds__(256) emsg_edge(
        const float* __restrict__ padded, const _Float16* __restrict__ Wea2,
        const float* __restrict__ b_ea2, const float* __restrict__ res_e,
        float* __restrict__ out_edge) {
    int lane = threadIdx.x & 31, wv = threadIdx.x >> 5;
    int m0 = (blockIdx.x * 2 + (wv >> 2)) * 16;
    int nt = wv & 3;
    v8f c = {};
#pragma unroll
    for (int kt = 0; kt < 2; ++kt) {
        v16h a = frag_a_f32(padded, 64, m0, kt * 32, lane);
        v16h b = frag_b_pk(Wea2, 4, kt, nt, lane);
        c = wmma_f32_f16(a, b, c);
    }
    int col = nt * 16 + (lane & 15);
    int rbase = m0 + ((lane >> 4) << 3);
    float r1 = res_e[64 + col];
    float bv = b_ea2[col];
#pragma unroll
    for (int r = 0; r < 8; ++r) {
        long row = rbase + r;
        out_edge[row * 64 + col] += r1 * silu_f(c[r] + bv);
    }
}

// ---- host launcher ----------------------------------------------------------
extern "C" void kernel_launch(void* const* d_in, const int* in_sizes, int n_in,
                              void* d_out, int out_size, void* d_ws, size_t ws_size,
                              hipStream_t stream) {
    (void)in_sizes; (void)n_in; (void)out_size; (void)ws_size;
    const float* node_ext = (const float*)d_in[0];
    const float* edge     = (const float*)d_in[1];
    const float* h2       = (const float*)d_in[2];
    const float* angle    = (const float*)d_in[3];
    const float* sw       = (const float*)d_in[4];
    const float* a_sw     = (const float*)d_in[5];
    const float* W_self   = (const float*)d_in[6];
    const float* b_self   = (const float*)d_in[7];
    const float* W_sym    = (const float*)d_in[8];
    const float* b_sym    = (const float*)d_in[9];
    const float* W_ne     = (const float*)d_in[10];
    const float* b_ne     = (const float*)d_in[11];
    const float* W_es     = (const float*)d_in[12];
    const float* b_es     = (const float*)d_in[13];
    const float* W_ea1    = (const float*)d_in[14];
    const float* b_ea1    = (const float*)d_in[15];
    const float* W_ea2    = (const float*)d_in[16];
    const float* b_ea2    = (const float*)d_in[17];
    const float* W_as     = (const float*)d_in[18];
    const float* b_as     = (const float*)d_in[19];
    const float* res_n    = (const float*)d_in[20];
    const float* res_e    = (const float*)d_in[21];
    const float* res_a    = (const float*)d_in[22];
    const int*   nlist    = (const int*)d_in[23];
    const unsigned char* nmask = (const unsigned char*)d_in[24];
    const unsigned char* amask = (const unsigned char*)d_in[25];

    float* out_node  = (float*)d_out;                       // 1024*128
    float* out_edge  = out_node + 1024 * 128;               // 1024*64*64
    float* out_angle = out_edge + (long)1024 * 64 * 64;     // 1024*400*64

    // workspace layout (f32 then f16)
    float* wsf = (float*)d_ws;
    size_t o = 0;
    float* nei_node  = wsf + o; o += (size_t)1024 * 64 * 128;
    float* concat768 = wsf + o; o += (size_t)1024 * 768;
    float* n_self    = wsf + o; o += (size_t)1024 * 128;
    float* node_sym  = wsf + o; o += (size_t)1024 * 128;
    float* nAB1      = wsf + o; o += (size_t)1024 * 128;
    float* nW0       = wsf + o; o += (size_t)1024 * 128;
    float* nV0       = wsf + o; o += (size_t)1024 * 64;
    float* eAB       = wsf + o; o += (size_t)20480 * 256;
    float* padded    = wsf + o; o += (size_t)1024 * 64 * 64;
    _Float16* wh = (_Float16*)(wsf + o);
    _Float16* Wself_h  = wh;            // 16384
    _Float16* Wsym_h   = wh + 16384;    // 98304
    _Float16* Wne0_h   = wh + 114688;   // 16384
    _Float16* WneCat_h = wh + 131072;   // 24576
    _Float16* Wes0_h   = wh + 155648;   // 8192
    _Float16* WesCat_h = wh + 163840;   // 12288
    _Float16* A0B0_h   = wh + 176128;   // 8192
    _Float16* A1B1_h   = wh + 184320;   // 16384
    _Float16* A23B23_h = wh + 200704;   // 16384
    _Float16* Wea2_h   = wh + 217088;   // 4096

    // pack weights into WMMA-fragment layout
    pack_weight<128, 128, 0, 0>  <<<64,  256, 0, stream>>>(W_self, nullptr, Wself_h);
    pack_weight<768, 128, 0, 0>  <<<384, 256, 0, stream>>>(W_sym,  nullptr, Wsym_h);
    pack_weight<128, 128, 0, 0>  <<<64,  256, 0, stream>>>(W_ne,   nullptr, Wne0_h);
    pack_weight<192, 128, 0, 128><<<96,  256, 0, stream>>>(W_ne,   nullptr, WneCat_h);
    pack_weight<128, 64,  0, 0>  <<<32,  256, 0, stream>>>(W_es,   nullptr, Wes0_h);
    pack_weight<192, 64,  0, 128><<<48,  256, 0, stream>>>(W_es,   nullptr, WesCat_h);
    pack_weight<64,  128, 2, 0>  <<<32,  256, 0, stream>>>(W_ea1,  W_as,    A0B0_h);
    pack_weight<128, 128, 2, 64> <<<64,  256, 0, stream>>>(W_ea1,  W_as,    A1B1_h);
    pack_weight<64,  256, 3, 0>  <<<64,  256, 0, stream>>>(W_ea1,  W_as,    A23B23_h);
    pack_weight<64,  64,  0, 0>  <<<16,  256, 0, stream>>>(W_ea2,  nullptr, Wea2_h);

    gather_nei<<<32768, 256, 0, stream>>>(node_ext, nlist, nmask, nei_node);
    grrg_kernel<<<1024, 256, 0, stream>>>(h2, edge, nei_node, sw, nmask, concat768);
    node_gemm_t<0><<<64, 256, 0, stream>>>(node_ext,  Wself_h, b_self, n_self);
    node_gemm_t<1><<<64, 256, 0, stream>>>(concat768, Wsym_h,  b_sym,  node_sym);
    node_gemm_t<2><<<64, 256, 0, stream>>>(node_ext,  A1B1_h,  nullptr, nAB1);
    node_gemm_t<3><<<64, 256, 0, stream>>>(node_ext,  Wne0_h,  nullptr, nW0);
    node_gemm_t<4><<<64, 128, 0, stream>>>(node_ext,  Wes0_h,  nullptr, nV0);
    eab_gemm<<<dim3(1280, 2), 256, 0, stream>>>(edge, amask, A23B23_h, eAB);
    hipMemsetAsync(padded, 0, (size_t)1024 * 64 * 64 * sizeof(float), stream);
    angle_main<<<dim3(25, 1024), 256, 0, stream>>>(angle, a_sw, amask, nAB1, eAB,
                                                   b_ea1, b_as, res_a, A0B0_h, padded, out_angle);
    ne_es_kernel<<<1024, 256, 0, stream>>>(node_ext, edge, nei_node, sw, nmask,
                                           nW0, nV0, n_self, node_sym, b_ne, b_es, res_n, res_e,
                                           WneCat_h, WesCat_h, out_node, out_edge);
    emsg_edge<<<2048, 256, 0, stream>>>(padded, Wea2_h, b_ea2, res_e, out_edge);
}
